// NanoPitch_66297115181520
// MI455X (gfx1250) — matrix operand order repaired and checked by reading, not compiled, not run on hardware
//
#include <hip/hip_runtime.h>
#include <hip/hip_bf16.h>

// ---------------------------------------------------------------------------
// NanoPitch on gfx1250 (wave32, WMMA):
//  - one-time weight pack kernels  (f32 -> bf16, conv taps de-interleaved)
//  - bf16 WMMA GEMMs (f32 accumulate) for convs / GRU input proj / pitch head
//  - persistent WMMA recurrence kernel for the 3 stacked GRUs, with all
//    gate-phase addressing hoisted out of the 4000-step serial loop
// ---------------------------------------------------------------------------

#define BB     32
#define TT     4000
#define NMELS  40
#define CONDC  64
#define HGRU   96
#define H3     288      // 3*HGRU
#define NPITCH 360
#define NCAT   384      // 4*HGRU
#define MT16   ((BB*TT)/16)   // 8000 row tiles of 16

typedef __attribute__((ext_vector_type(16))) __bf16 v16bf;
typedef __attribute__((ext_vector_type(8)))  __bf16 bf16x8;
typedef __attribute__((ext_vector_type(8)))  float  v8f;
typedef __attribute__((ext_vector_type(4)))  float  f32x4;

__device__ __forceinline__ float fsigmoid(float x) { return 1.0f / (1.0f + __expf(-x)); }
__device__ __forceinline__ float ftanh(float x)    { return 2.0f / (1.0f + __expf(-2.0f * x)) - 1.0f; }

// Convert an 8-float run (two float4) into elements [o..o+7] of an A fragment.
__device__ __forceinline__ void cvt_run(v16bf& a, int o, f32x4 p0, f32x4 p1) {
  a[o + 0] = (__bf16)p0[0]; a[o + 1] = (__bf16)p0[1];
  a[o + 2] = (__bf16)p0[2]; a[o + 3] = (__bf16)p0[3];
  a[o + 4] = (__bf16)p1[0]; a[o + 5] = (__bf16)p1[1];
  a[o + 6] = (__bf16)p1[2]; a[o + 7] = (__bf16)p1[3];
}

// ---------------------------------------------------------------------------
// One-time weight packing (weights are reused by 8000 M-tiles; convert once).
// ---------------------------------------------------------------------------
__global__ void pack_conv_w(const float* __restrict__ W, int Cin, int Cout,
                            __bf16* __restrict__ Wp) {
  const int idx = blockIdx.x * 256 + threadIdx.x;
  if (idx >= Cout * 192) return;
  const int o = idx / 192, r = idx % 192, tap = r / 64, c = r % 64;
  const float v = (c < Cin) ? W[((size_t)o * Cin + c) * 3 + tap] : 0.0f;
  Wp[idx] = (__bf16)v;
}

__global__ void pack_w(const float* __restrict__ W, int total, __bf16* __restrict__ Wp) {
  const int idx = blockIdx.x * 256 + threadIdx.x;
  if (idx < total) Wp[idx] = (__bf16)W[idx];
}

// ---------------------------------------------------------------------------
// Causal conv (kernel 3, left-pad 2) as tap-accumulated GEMM, NT col-tiles per
// wave (A fragment reused NT times). M tiles never straddle batches (T%16==0).
// A lane L: row = L&15 ; K = 16*(L>>4) + {0..15}        (contiguous k order)
// B lane L: col = L&15 ; K = 16*(L>>4) + {0..15}        (same order as A)
// ---------------------------------------------------------------------------
template <int NT>
__global__ void conv_wmma_kernel(const float* __restrict__ X, int Cin,
                                 const __bf16* __restrict__ Wp,   // (Cout,192)
                                 const float* __restrict__ bias,
                                 float* __restrict__ Y, int ldY) {
  const int lane = threadIdx.x & 31;
  const int m0 = blockIdx.x * 16;
  const int mr = m0 + (lane & 15);
  const int t  = mr % TT;
  const int kg = (lane >> 4) * 8;
  const int kh = (lane >> 4) * 16;

  int nn[NT]; const __bf16* wr[NT];
#pragma unroll
  for (int i = 0; i < NT; ++i) {
    nn[i] = (blockIdx.y * NT + i) * 16 + (lane & 15);   // always < Cout here
    wr[i] = Wp + (size_t)nn[i] * 192;
  }
  v8f c[NT] = {};

  for (int tap = 0; tap < 3; ++tap) {
    const bool trow = (t + tap - 2) >= 0;                 // causal boundary
    const float* xr = X + (size_t)(mr + tap - 2) * Cin;   // same batch shift
#pragma unroll
    for (int kc2 = 0; kc2 < 2; ++kc2) {                   // CinPad = 64
      const int cb = kc2 * 32;
      v16bf a = {};
#pragma unroll
      for (int r = 0; r < 2; ++r) {                       // two 8-elem runs
        const int base = cb + kh + r * 8;
        if (trow && base < Cin) {                         // 8 | Cin -> run-uniform
          f32x4 p0 = *(const f32x4*)(xr + base);
          f32x4 p1 = *(const f32x4*)(xr + base + 4);
          cvt_run(a, r * 8, p0, p1);
        }
      }
      const int kb = tap * 64 + cb + kh;                  // 32B-aligned
#pragma unroll
      for (int i = 0; i < NT; ++i) {
        v16bf b = *(const v16bf*)(wr[i] + kb);
        c[i] = __builtin_amdgcn_wmma_f32_16x16x32_bf16(false, a, false, b,
                                                       (short)0, c[i], false, false);
      }
    }
  }
#pragma unroll
  for (int i = 0; i < NT; ++i) {
    const float bn = bias[nn[i]];
#pragma unroll
    for (int v = 0; v < 8; ++v)
      Y[(size_t)(m0 + v + kg) * ldY + nn[i]] = ftanh(c[i][v] + bn);
  }
}

// ---------------------------------------------------------------------------
// Y[M,N] = act(X[M,K] @ Wp[N,K]^T + b), K%32==0, Wp pre-packed bf16.
// NT column tiles per wave; partial last tile handled by clamped B rows and
// predicated stores (WMMA always executes with full EXEC).
// ACT: 0 = identity (GRU input projection), 2 = sigmoid (pitch head).
// ---------------------------------------------------------------------------
template <int ACT, int NT>
__global__ void gemm_wmma_kernel(const float* __restrict__ X, int ldX, int K,
                                 const __bf16* __restrict__ Wp, int N,
                                 const float* __restrict__ bias,
                                 float* __restrict__ Y, int ldY) {
  const int lane = threadIdx.x & 31;
  const int m0 = blockIdx.x * 16;
  const int kg = (lane >> 4) * 8;
  const int kh = (lane >> 4) * 16;
  const float* xr = X + (size_t)(m0 + (lane & 15)) * ldX;

  int n[NT]; bool nv[NT]; const __bf16* wr[NT];
#pragma unroll
  for (int i = 0; i < NT; ++i) {
    n[i]  = (blockIdx.y * NT + i) * 16 + (lane & 15);
    nv[i] = n[i] < N;
    wr[i] = Wp + (size_t)(nv[i] ? n[i] : N - 1) * K;
  }
  v8f c[NT] = {};

  for (int kc = 0; kc < K; kc += 32) {
    f32x4 p0 = *(const f32x4*)(xr + kc + kh);
    f32x4 p1 = *(const f32x4*)(xr + kc + kh + 4);
    f32x4 p2 = *(const f32x4*)(xr + kc + kh + 8);
    f32x4 p3 = *(const f32x4*)(xr + kc + kh + 12);
    v16bf a;
    cvt_run(a, 0, p0, p1);
    cvt_run(a, 8, p2, p3);
#pragma unroll
    for (int i = 0; i < NT; ++i) {
      v16bf b = *(const v16bf*)(wr[i] + kc + kh);
      c[i] = __builtin_amdgcn_wmma_f32_16x16x32_bf16(false, a, false, b,
                                                     (short)0, c[i], false, false);
    }
  }
#pragma unroll
  for (int i = 0; i < NT; ++i) {
    const float bn = nv[i] ? bias[n[i]] : 0.0f;
#pragma unroll
    for (int v = 0; v < 8; ++v) {
      float val = c[i][v] + bn;
      if (ACT == 2) val = fsigmoid(val);
      if (nv[i]) Y[(size_t)(m0 + v + kg) * ldY + n[i]] = val;
    }
  }
}

// ---------------------------------------------------------------------------
// Persistent GRU recurrence. One block per 16-batch group (grid = 2).
// 18 waves: wave w owns output columns 16w..16w+15 of hp = h @ whh^T + bhh,
// whh held as bf16 B-fragments in registers for the whole 4000-step sequence.
// h lives in LDS (f32 for gates + bf16 row-major copy gathered as A fragments
// via two 16B ds_load_b128 per 32-wide K chunk). 3 chained v_wmma per step
// (vs 24 with 16x16x4_f32) minimizes the serial accumulator dependency.
// Gate-phase (mm,j) decomposition, LDS offsets, and global base pointers are
// all time-invariant per thread -> hoisted out of the step loop; per step the
// only address work is xp += H3, gout += NCAT. Step t+1's xp row prefetched.
// ---------------------------------------------------------------------------
__global__ __launch_bounds__(576, 1)
void gru_rec_kernel(const float* __restrict__ xp,      // (B, T, 288)
                    const float* __restrict__ whh,     // (288, 96)
                    const float* __restrict__ bhh,     // (288)
                    float* __restrict__ gout,          // cat + 96*(l+1), ld NCAT
                    float* __restrict__ hout) {        // (B, 96) slice
  __shared__ __align__(16) __bf16 hA[16][HGRU];
  __shared__ float hF[16][HGRU];
  __shared__ float hp[16][H3];

  const int tid  = threadIdx.x;          // 0..575 (18 waves)
  const int lane = tid & 31;
  const int wave = tid >> 5;
  const int bg   = blockIdx.x;           // batch group
  const int col  = wave * 16 + (lane & 15);
  const int m    = lane & 15;
  const int kg   = (lane >> 4) * 8;
  const int kh   = (lane >> 4) * 16;

  // Preload whh^T B-fragments (3 K-chunks of 32) into registers, bf16.
  v16bf Bt[3];
  {
    const float* wr = whh + (size_t)col * HGRU;
#pragma unroll
    for (int kc = 0; kc < 3; ++kc) {
      v16bf b;
#pragma unroll
      for (int e = 0; e < 16; ++e) b[e] = (__bf16)wr[kc * 32 + kg + e + (e & 8)];
      Bt[kc] = b;
    }
  }
  const float bias = bhh[col];

  // Time-invariant gate-phase setup: thread tid owns units {tid, tid+576,
  // tid+1152 (if tid<384)} of the 16x96 state update.
  const int nu = (tid < 1536 - 2 * 576) ? 3 : 2;
  int hpo[3], hfo[3];                 // LDS float offsets (fixed)
  const float* xru[3];                // advances by H3 per step
  float* gou[3];                      // advances by NCAT per step
#pragma unroll
  for (int u = 0; u < 3; ++u) {
    const int idx = tid + u * 576;
    const int mm = (idx < 1536) ? (idx / HGRU) : 0;
    const int j  = (idx < 1536) ? (idx - mm * HGRU) : 0;
    hpo[u] = mm * H3 + j;
    hfo[u] = mm * HGRU + j;
    xru[u] = xp + (size_t)(bg * 16 + mm) * TT * H3 + j;
    gou[u] = gout + (size_t)(bg * 16 + mm) * TT * NCAT + j;
  }

  for (int i = tid; i < 16 * HGRU; i += 576) {
    ((float*)hF)[i]  = 0.0f;
    ((__bf16*)hA)[i] = (__bf16)0.0f;
  }
  __syncthreads();

  for (int t = 0; t < TT; ++t) {
    // hp tile = h(16x96) @ whh^T(96x16) + bhh
    v8f c = {};
#pragma unroll
    for (int kc = 0; kc < 3; ++kc) {
      bf16x8 lo = *(const bf16x8*)&hA[m][kc * 32 + kg];
      bf16x8 hi = *(const bf16x8*)&hA[m][kc * 32 + kg + 16];
      v16bf a = __builtin_shufflevector(lo, hi, 0, 1, 2, 3, 4, 5, 6, 7,
                                        8, 9, 10, 11, 12, 13, 14, 15);
      c = __builtin_amdgcn_wmma_f32_16x16x32_bf16(false, a, false, Bt[kc],
                                                  (short)0, c, false, false);
    }
#pragma unroll
    for (int v = 0; v < 8; ++v) hp[v + kg][col] = c[v] + bias;  // D row = v+8*(L>>4)
    __syncthreads();

    // Gates: 16 batches x 96 hidden = 1536 units over 576 threads.
#pragma unroll
    for (int u = 0; u < 3; ++u) {
      if (u < nu) {
        const float* xr = xru[u];
        __builtin_prefetch(xr + H3, 0, 1);     // next step's input row
        const float* hpp = (const float*)hp + hpo[u];
        const float r  = fsigmoid(xr[0] + hpp[0]);
        const float z  = fsigmoid(xr[HGRU] + hpp[HGRU]);
        const float nn = ftanh(xr[2 * HGRU] + r * hpp[2 * HGRU]);
        const float hn = (1.0f - z) * nn + z * ((const float*)hF)[hfo[u]];
        ((float*)hF)[hfo[u]]  = hn;
        ((__bf16*)hA)[hfo[u]] = (__bf16)hn;
        *gou[u] = hn;
        xru[u] += H3;
        gou[u] += NCAT;
      }
    }
    __syncthreads();
  }

  for (int idx = tid; idx < 16 * HGRU; idx += 576) {
    const int mm = idx / HGRU;
    const int j  = idx - mm * HGRU;
    hout[(size_t)(bg * 16 + mm) * HGRU + j] = hF[mm][j];
  }
}

// ---------------------------------------------------------------------------
// vad head: one wave per row, 384-length dot, wave32 xor-shuffle reduction.
// ---------------------------------------------------------------------------
__global__ void vad_kernel(const float* __restrict__ cat, const float* __restrict__ vw,
                           const float* __restrict__ vb, float* __restrict__ out) {
  const int lane = threadIdx.x & 31;
  const int wid  = (int)((blockIdx.x * blockDim.x + threadIdx.x) >> 5);
  float w[12];
#pragma unroll
  for (int i = 0; i < 12; ++i) w[i] = vw[lane + 32 * i];
  const float* r = cat + (size_t)wid * NCAT;
  float s = 0.0f;
#pragma unroll
  for (int i = 0; i < 12; ++i) s += r[lane + 32 * i] * w[i];
#pragma unroll
  for (int ofs = 16; ofs >= 1; ofs >>= 1) s += __shfl_xor(s, ofs, 32);
  if (lane == 0) out[wid] = fsigmoid(s + vb[0]);
}

// ---------------------------------------------------------------------------
extern "C" void kernel_launch(void* const* d_in, const int* in_sizes, int n_in,
                              void* d_out, int out_size, void* d_ws, size_t ws_size,
                              hipStream_t stream) {
  (void)in_sizes; (void)n_in; (void)out_size; (void)ws_size;

  const float* mel = (const float*)d_in[0];
  const float* c1w = (const float*)d_in[1];
  const float* c1b = (const float*)d_in[2];
  const float* c2w = (const float*)d_in[3];
  const float* c2b = (const float*)d_in[4];
  const float* wih[3] = {(const float*)d_in[5],  (const float*)d_in[9],  (const float*)d_in[13]};
  const float* whh[3] = {(const float*)d_in[6],  (const float*)d_in[10], (const float*)d_in[14]};
  const float* bih[3] = {(const float*)d_in[7],  (const float*)d_in[11], (const float*)d_in[15]};
  const float* bhh[3] = {(const float*)d_in[8],  (const float*)d_in[12], (const float*)d_in[16]};
  const float* vadw = (const float*)d_in[17];
  const float* vadb = (const float*)d_in[18];
  const float* pw   = (const float*)d_in[19];
  const float* pb   = (const float*)d_in[20];

  float* out       = (float*)d_out;
  float* vad_out   = out;                                  // (B,T,1)
  float* pitch_out = out + (size_t)BB * TT;                // (B,T,360)
  float* h_out     = pitch_out + (size_t)BB * TT * NPITCH; // h1,h2,h3

  char* ws = (char*)d_ws;
  size_t off = 0;
  float* cat = (float*)(ws + off); off += (size_t)BB * TT * NCAT * 4;   // (B,T,384)
  float* xp  = (float*)(ws + off); off += (size_t)BB * TT * H3 * 4;     // (B,T,288)
  float* c1  = (float*)(ws + off); off += (size_t)BB * TT * CONDC * 4;  // (B,T,64)
  __bf16* w1p = (__bf16*)(ws + off); off += (size_t)CONDC * 192 * 2;    // packed conv1 w
  __bf16* w2p = (__bf16*)(ws + off); off += (size_t)HGRU * 192 * 2;     // packed conv2 w
  __bf16* wihp[3];
  for (int l = 0; l < 3; ++l) { wihp[l] = (__bf16*)(ws + off); off += (size_t)H3 * HGRU * 2; }
  __bf16* pwp = (__bf16*)(ws + off); off += (size_t)NPITCH * NCAT * 2;

  // --- one-time weight packing (f32 -> bf16, conv taps de-interleaved) ---
  pack_conv_w<<<dim3((CONDC * 192 + 255) / 256), 256, 0, stream>>>(c1w, NMELS, CONDC, w1p);
  pack_conv_w<<<dim3((HGRU  * 192 + 255) / 256), 256, 0, stream>>>(c2w, CONDC, HGRU, w2p);
  for (int l = 0; l < 3; ++l)
    pack_w<<<dim3((H3 * HGRU + 255) / 256), 256, 0, stream>>>(wih[l], H3 * HGRU, wihp[l]);
  pack_w<<<dim3((NPITCH * NCAT + 255) / 256), 256, 0, stream>>>(pw, NPITCH * NCAT, pwp);

  // conv1: mel(40) -> c1(64), tanh          (4 tiles, NT=2 -> grid.y=2)
  conv_wmma_kernel<2><<<dim3(MT16, 2), 32, 0, stream>>>(mel, NMELS, w1p, c1b, c1, CONDC);
  // conv2: c1(64) -> cat cols [0:96), tanh  (6 tiles, NT=3 -> grid.y=2)
  conv_wmma_kernel<3><<<dim3(MT16, 2), 32, 0, stream>>>(c1, CONDC, w2p, c2b, cat, NCAT);

  // Stacked GRUs: parallel input projection, then persistent recurrence.
  for (int l = 0; l < 3; ++l) {
    gemm_wmma_kernel<0, 3><<<dim3(MT16, 6), 32, 0, stream>>>(
        cat + 96 * l, NCAT, HGRU, wihp[l], H3, bih[l], xp, H3);
    gru_rec_kernel<<<2, 576, 0, stream>>>(
        xp, whh[l], bhh[l], cat + 96 * (l + 1), h_out + (size_t)l * BB * HGRU);
  }

  // pitch head: cat(384) -> 360, sigmoid (23 tiles, NT=3 -> grid.y=8, masked)
  gemm_wmma_kernel<2, 3><<<dim3(MT16, 8), 32, 0, stream>>>(
      cat, NCAT, NCAT, pwp, NPITCH, pb, pitch_out, NPITCH);
  // vad head
  vad_kernel<<<dim3((BB * TT) / 8), 256, 0, stream>>>(cat, vadw, vadb, vad_out);
}
